// MVSC_Commonality_Decoder_47991964566152
// MI455X (gfx1250) — matrix operand order repaired and checked by reading, not compiled, not run on hardware
//
#include <hip/hip_runtime.h>
#include <hip/hip_bf16.h>
#include <math.h>

typedef __bf16 bf16t;
typedef __attribute__((ext_vector_type(16))) __bf16 v16bf;
typedef __attribute__((ext_vector_type(8)))  float  v8f;

union Frag16 { v16bf v; uint4 q[2]; };
union Pack8  { bf16t h[8]; uint4 u; };
union Pack4  { bf16t h[4]; uint2 u; };

// Async global->LDS copy helpers (CDNA5 GLOBAL_LOAD_ASYNC_TO_LDS_*, ASYNCcnt).
// Generic pointers to __shared__ carry the LDS byte address in their low 32 bits.
__device__ __forceinline__ void async_copy_b128x2(void* lds, const void* gptr) {
  const uint32_t la = (uint32_t)(size_t)lds;
  const uint64_t ga = (uint64_t)(uintptr_t)gptr;
  asm volatile(
      "global_load_async_to_lds_b128 %0, %1, off\n\t"
      "global_load_async_to_lds_b128 %0, %1, off offset:16" ::"v"(la),
      "v"(ga)
      : "memory");
}
__device__ __forceinline__ void async_copy_b64(void* lds, const void* gptr) {
  const uint32_t la = (uint32_t)(size_t)lds;
  const uint64_t ga = (uint64_t)(uintptr_t)gptr;
  asm volatile("global_load_async_to_lds_b64 %0, %1, off" ::"v"(la), "v"(ga)
               : "memory");
}
__device__ __forceinline__ void async_wait0() {
  asm volatile("s_wait_asynccnt 0x0" ::: "memory");
}

// ---------------------------------------------------------------------------
// WMMA helpers (gfx1250, wave32). D = A(16x32 bf16) * B(32x16 bf16) + C(f32)
// A layout: lane = M row (lane&15); per lane K = {hi*8..hi*8+7, 16+hi*8..+7}
//           -> two contiguous 16B runs.
// B layout: lane = N col (lane&15); per lane K = {hi*8..hi*8+7, 16+hi*8..+7}
//           -> two contiguous 16B runs.
// ---------------------------------------------------------------------------
__device__ __forceinline__ v8f wmma_bf16(v16bf a, v16bf b, v8f c) {
  return __builtin_amdgcn_wmma_f32_16x16x32_bf16(false, a, false, b, (short)0, c,
                                                 false, false);
}

// Row-major source, row = M row (A operand). ld multiple of 8, k0 multiple of 8.
__device__ __forceinline__ v16bf load_fragA_vec(const bf16t* p, int row, int ld,
                                                int k0, int lane) {
  const int hi = lane >> 4;
  const bf16t* q = p + (size_t)row * ld + k0 + hi * 8;
  Frag16 f;
  f.q[0] = *(const uint4*)(q);
  f.q[1] = *(const uint4*)(q + 16);
  return f.v;
}

// Source where each lane's column is K-contiguous (i.e. X^T stored row-major,
// row index = N column of B). ld multiple of 8, k0 multiple of 8.
__device__ __forceinline__ v16bf load_fragB_vec(const bf16t* p, int row, int ld,
                                                int k0, int lane) {
  const int hi = lane >> 4;
  const bf16t* q = p + (size_t)row * ld + k0 + hi * 8;
  Frag16 f;
  f.q[0] = *(const uint4*)(q);
  f.q[1] = *(const uint4*)(q + 16);
  return f.v;
}

// ---------------------------------------------------------------------------
// Generic WMMA GEMM: C[M,N] = A(bf16)[M,K] * Bw(f32->bf16)[K,N] (+bias)(+GELU)
// Block tile 128x64, 8 waves of 32x32. M%128==0, N%64==0, K%32==0.
// A tile staged with async global->LDS (no VGPR round-trip, overlaps with the
// f32->bf16 weight-tile staging). Weight tile stored TRANSPOSED in LDS so B
// fragments are b128 loads.
// mode: 0 = f32 out, 1 = bf16 out, 2 = GELU + bf16 out
// ---------------------------------------------------------------------------
__global__ __launch_bounds__(256) void gemm_bf16(
    const bf16t* __restrict__ A, int lda, const float* __restrict__ Bw, int ldb,
    const float* __restrict__ bias, float* __restrict__ Cf, bf16t* __restrict__ Cb,
    int ldc, int K, int mode) {
  __shared__ __align__(16) bf16t shA[128 * 40];
  __shared__ __align__(16) bf16t shBT[64 * 40];  // [n][k]
  const int lane = threadIdx.x & 31;
  const int wave = threadIdx.x >> 5;
  const int hi = lane >> 4;
  const int lo = lane & 15;
  const int wm = wave >> 1;  // 0..3 -> M quadrant (32 rows)
  const int wn = wave & 1;   // 0..1 -> N half (32 cols)
  const size_t m0 = (size_t)blockIdx.y * 128;
  const int n0 = blockIdx.x * 64;
  const int ar = threadIdx.x >> 1, ah = (threadIdx.x & 1) * 16;

  v8f acc[2][2];
#pragma unroll
  for (int a = 0; a < 2; ++a)
#pragma unroll
    for (int b = 0; b < 2; ++b) acc[a][b] = (v8f){0, 0, 0, 0, 0, 0, 0, 0};

  for (int kt = 0; kt < K; kt += 32) {
    __syncthreads();
    // A tile: 128x32 bf16, 32B per thread, async DMA into LDS
    async_copy_b128x2(&shA[ar * 40 + ah], A + (m0 + ar) * lda + kt + ah);
    // B tile: 32x64 f32 -> bf16, stored transposed; column-wise coalesced loads
#pragma unroll
    for (int c = threadIdx.x; c < 512; c += 256) {
      const int n = c & 63, r4 = (c >> 6) * 4;
      Pack4 pk;
#pragma unroll
      for (int j = 0; j < 4; ++j)
        pk.h[j] = (bf16t)Bw[(size_t)(kt + r4 + j) * ldb + n0 + n];
      *(uint2*)&shBT[n * 40 + r4] = pk.u;
    }
    if (kt + 32 < K)  // CDNA5 global_prefetch_b8 of next A tile
      __builtin_prefetch(A + (m0 + ar) * lda + kt + 32, 0, 1);
    async_wait0();
    __syncthreads();

    v16bf af[2], bfr[2];
#pragma unroll
    for (int mt = 0; mt < 2; ++mt)
      af[mt] = load_fragA_vec(shA, wm * 32 + mt * 16 + lo, 40, 0, lane);
#pragma unroll
    for (int nt = 0; nt < 2; ++nt)
      bfr[nt] = load_fragB_vec(shBT, wn * 32 + nt * 16 + lo, 40, 0, lane);
#pragma unroll
    for (int mt = 0; mt < 2; ++mt)
#pragma unroll
      for (int nt = 0; nt < 2; ++nt)
        acc[mt][nt] = wmma_bf16(af[mt], bfr[nt], acc[mt][nt]);
  }

#pragma unroll
  for (int mt = 0; mt < 2; ++mt)
#pragma unroll
    for (int nt = 0; nt < 2; ++nt)
#pragma unroll
      for (int v = 0; v < 8; ++v) {
        const size_t row = m0 + wm * 32 + mt * 16 + hi * 8 + v;
        const int col = n0 + wn * 32 + nt * 16 + lo;
        float xv = acc[mt][nt][v];
        if (bias) xv += bias[col];
        if (mode == 2) xv = 0.5f * xv * (1.0f + erff(xv * 0.70710678118654752f));
        if (mode == 0)
          Cf[row * ldc + col] = xv;
        else
          Cb[row * ldc + col] = (bf16t)xv;
      }
}

// ---------------------------------------------------------------------------
// LayerNorm (C=256) with optional shift + 3D window partition (WS=2,8,8),
// H=W=64. One wave per token, vectorized float4 loads / b128 bf16 store.
// ---------------------------------------------------------------------------
__global__ __launch_bounds__(256) void ln_part(
    const float* __restrict__ h, const float* __restrict__ g,
    const float* __restrict__ bb, bf16t* __restrict__ out, int winmode, int D,
    int sd, int sh, int sw) {
  const int wave = threadIdx.x >> 5, lane = threadIdx.x & 31;
  const int tk = blockIdx.x * 8 + wave;
  size_t src;
  if (winmode) {
    const int t = tk & 127, win = tk >> 7;
    const int ww = win & 7, wh = (win >> 3) & 7;
    const int Dw = D >> 1;
    const int wd = (win >> 6) % Dw, b = (win >> 6) / Dw;
    const int cd = t >> 6, ch = (t >> 3) & 7, cw = t & 7;
    const int dss = (wd * 2 + cd + sd) % D;
    const int hss = (wh * 8 + ch + sh) & 63;
    const int wss = (ww * 8 + cw + sw) & 63;
    src = (size_t)(b * D + dss) * 4096 + hss * 64 + wss;
  } else {
    src = (size_t)tk;
  }
  const float* xp = h + src * 256 + lane * 8;
  union { float4 f4[2]; float f[8]; } x, gg, bv;
  x.f4[0] = *(const float4*)(xp);
  x.f4[1] = *(const float4*)(xp + 4);
  float s = 0.f;
#pragma unroll
  for (int j = 0; j < 8; ++j) s += x.f[j];
#pragma unroll
  for (int o = 16; o > 0; o >>= 1) s += __shfl_xor(s, o, 32);
  const float mu = s * (1.f / 256.f);
  float vv = 0.f;
#pragma unroll
  for (int j = 0; j < 8; ++j) {
    const float d = x.f[j] - mu;
    vv += d * d;
  }
#pragma unroll
  for (int o = 16; o > 0; o >>= 1) vv += __shfl_xor(vv, o, 32);
  const float rstd = rsqrtf(vv * (1.f / 256.f) + 1e-5f);
  gg.f4[0] = *(const float4*)(g + lane * 8);
  gg.f4[1] = *(const float4*)(g + lane * 8 + 4);
  bv.f4[0] = *(const float4*)(bb + lane * 8);
  bv.f4[1] = *(const float4*)(bb + lane * 8 + 4);
  Pack8 o8;
#pragma unroll
  for (int j = 0; j < 8; ++j)
    o8.h[j] = (bf16t)((x.f[j] - mu) * rstd * gg.f[j] + bv.f[j]);
  *(uint4*)&out[(size_t)tk * 256 + lane * 8] = o8.u;
}

// Window reverse + unshift + residual add: h[dst] += y[win order]
__global__ __launch_bounds__(256) void win_reverse_add(
    const float* __restrict__ y, float* __restrict__ h, int D, int sd, int sh,
    int sw) {
  const int wave = threadIdx.x >> 5, lane = threadIdx.x & 31;
  const int tk = blockIdx.x * 8 + wave;
  const int t = tk & 127, win = tk >> 7;
  const int ww = win & 7, wh = (win >> 3) & 7;
  const int Dw = D >> 1;
  const int wd = (win >> 6) % Dw, b = (win >> 6) / Dw;
  const int cd = t >> 6, ch = (t >> 3) & 7, cw = t & 7;
  const int dss = (wd * 2 + cd + sd) % D;
  const int hss = (wh * 8 + ch + sh) & 63;
  const int wss = (ww * 8 + cw + sw) & 63;
  const float4* ys = (const float4*)(y + (size_t)tk * 256 + lane * 8);
  float4* hd = (float4*)(h +
      ((size_t)(b * D + dss) * 4096 + hss * 64 + wss) * 256 + lane * 8);
  float4 a0 = ys[0], a1 = ys[1];
  float4 b0 = hd[0], b1 = hd[1];
  b0.x += a0.x; b0.y += a0.y; b0.z += a0.z; b0.w += a0.w;
  b1.x += a1.x; b1.y += a1.y; b1.z += a1.z; b1.w += a1.w;
  hd[0] = b0; hd[1] = b1;
}

// ---------------------------------------------------------------------------
// Windowed attention: one (window, head) per 256-thread block.
// qkv bf16 rows of 768 (q|k|v each 4 heads x 64). NWIN=128, dh=64.
// Q/K staged via async global->LDS b64 (pure copy); V transposed via VGPRs.
// Q/K row-major (pitch 72), V^T (pitch 136), P row-major (pitch 136):
// every WMMA fragment is pure b128 LDS loads.
// ---------------------------------------------------------------------------
__global__ __launch_bounds__(256) void win_attn(const bf16t* __restrict__ qkv,
                                                const float* __restrict__ rpb,
                                                bf16t* __restrict__ out, int D,
                                                int shifted) {
  __shared__ __align__(16) bf16t shU[18432];     // Q [0,9216) K [9216,18432); then P [128][136]
  __shared__ __align__(16) bf16t shVT[64 * 136]; // V^T: [dh][token]
  const int head = blockIdx.x & 3;
  const int win = blockIdx.x >> 2;
  const int lane = threadIdx.x & 31;
  const int wave = threadIdx.x >> 5;
  const int hi = lane >> 4, lo = lane & 15;

  const size_t qbase = (size_t)win * 128 * 768 + head * 64;
  for (int i = threadIdx.x; i < 2048; i += 256) {
    const int t = i >> 4, c4 = (i & 15) * 4;
    const bf16t* row = qkv + qbase + (size_t)t * 768;
    async_copy_b64(&shU[t * 72 + c4], row + c4);               // q
    async_copy_b64(&shU[9216 + t * 72 + c4], row + 256 + c4);  // k
    Pack4 vv;
    vv.u = *(const uint2*)(row + 512 + c4);                    // v
#pragma unroll
    for (int j = 0; j < 4; ++j) shVT[(c4 + j) * 136 + t] = vv.h[j];
  }
  async_wait0();
  __syncthreads();

  const int r0 = wave * 16;  // this wave's 16 query rows
  v8f sc[8];
#pragma unroll
  for (int n = 0; n < 8; ++n) sc[n] = (v8f){0, 0, 0, 0, 0, 0, 0, 0};

  v16bf aq[2];
#pragma unroll
  for (int kk = 0; kk < 2; ++kk)
    aq[kk] = load_fragA_vec(shU, r0 + lo, 72, kk * 32, lane);
#pragma unroll
  for (int n = 0; n < 8; ++n) {
#pragma unroll
    for (int kk = 0; kk < 2; ++kk) {
      v16bf bk = load_fragB_vec(shU + 9216, n * 16 + lo, 72, kk * 32, lane);
      sc[n] = wmma_bf16(aq[kk], bk, sc[n]);
    }
  }

  // scale + relative position bias + shifted-window mask + softmax (per row)
  const int ww = win & 7, wh = (win >> 3) & 7;
  const int Dw = D >> 1;
  const int wd = (win >> 6) % Dw;
#pragma unroll
  for (int v = 0; v < 8; ++v) {
    const int tq = r0 + hi * 8 + v;
    const int cdq = tq >> 6, chq = (tq >> 3) & 7, cwq = tq & 7;
    int lq = 0;
    if (shifted) {
      const int pd = wd * 2 + cdq, ph = wh * 8 + chq, pw = ww * 8 + cwq;
      const int a = (pd < D - 2) ? 0 : ((pd == D - 2) ? 1 : 2);
      const int b2 = (ph < 56) ? 0 : ((ph < 60) ? 1 : 2);
      const int c2 = (pw < 56) ? 0 : ((pw < 60) ? 1 : 2);
      lq = a * 9 + b2 * 3 + c2;
    }
    float mx = -3.0e38f;
#pragma unroll
    for (int n = 0; n < 8; ++n) {
      const int tk2 = n * 16 + lo;
      const int cdk = tk2 >> 6, chk = (tk2 >> 3) & 7, cwk = tk2 & 7;
      const int rel = ((cdq - cdk + 1) * 15 + (chq - chk + 7)) * 15 + (cwq - cwk + 7);
      float xv = sc[n][v] * 0.125f + rpb[rel * 4 + head];
      if (shifted) {
        const int pd = wd * 2 + cdk, ph = wh * 8 + chk, pw = ww * 8 + cwk;
        const int a = (pd < D - 2) ? 0 : ((pd == D - 2) ? 1 : 2);
        const int b2 = (ph < 56) ? 0 : ((ph < 60) ? 1 : 2);
        const int c2 = (pw < 56) ? 0 : ((pw < 60) ? 1 : 2);
        if (a * 9 + b2 * 3 + c2 != lq) xv -= 100.0f;
      }
      sc[n][v] = xv;
      mx = fmaxf(mx, xv);
    }
#pragma unroll
    for (int o = 8; o > 0; o >>= 1) mx = fmaxf(mx, __shfl_xor(mx, o, 32));
    float sum = 0.f;
#pragma unroll
    for (int n = 0; n < 8; ++n) {
      const float e = __expf(sc[n][v] - mx);
      sc[n][v] = e;
      sum += e;
    }
#pragma unroll
    for (int o = 8; o > 0; o >>= 1) sum += __shfl_xor(sum, o, 32);
    const float inv = 1.f / sum;
#pragma unroll
    for (int n = 0; n < 8; ++n) sc[n][v] *= inv;
  }

  __syncthreads();  // all waves done reading Q/K region
#pragma unroll
  for (int v = 0; v < 8; ++v) {
    const int rr = r0 + hi * 8 + v;
#pragma unroll
    for (int n = 0; n < 8; ++n) shU[rr * 136 + n * 16 + lo] = (bf16t)sc[n][v];
  }
  __syncthreads();

  // out[16,64] = P[16,128] * V[128,64]
  v8f of[4];
#pragma unroll
  for (int n = 0; n < 4; ++n) of[n] = (v8f){0, 0, 0, 0, 0, 0, 0, 0};
#pragma unroll
  for (int kk = 0; kk < 4; ++kk) {
    v16bf pa = load_fragA_vec(shU, r0 + lo, 136, kk * 32, lane);
#pragma unroll
    for (int n = 0; n < 4; ++n) {
      v16bf vb = load_fragB_vec(shVT, n * 16 + lo, 136, kk * 32, lane);
      of[n] = wmma_bf16(pa, vb, of[n]);
    }
  }
  const size_t ob = ((size_t)win * 128) * 256 + head * 64;
#pragma unroll
  for (int n = 0; n < 4; ++n)
#pragma unroll
    for (int v = 0; v < 8; ++v) {
      const int rr = r0 + hi * 8 + v;
      out[ob + (size_t)rr * 256 + n * 16 + lo] = (bf16t)of[n][v];
    }
}

// ---------------------------------------------------------------------------
// Patch expand reorder + LayerNorm. src = [rows, 512] f32 (h @ exp_w),
// mode 0: view (V->2V), mode 1: temporal (T->2T, V = current V2). L=4096.
// ---------------------------------------------------------------------------
__global__ __launch_bounds__(256) void expand_ln(const float* __restrict__ src,
                                                 const float* __restrict__ g,
                                                 const float* __restrict__ b,
                                                 float* __restrict__ hout, int T,
                                                 int V, int mode) {
  const int L = 4096;
  const int wave = threadIdx.x >> 5, lane = threadIdx.x & 31;
  const int o = blockIdx.x * 8 + wave;
  const int l = o & (L - 1);
  int r = o >> 12;
  int e;
  size_t srow;
  if (mode == 0) {
    const int v2 = r % (2 * V);
    const int bt = r / (2 * V);  // b*T + t
    e = v2 & 1;
    const int v = v2 >> 1;
    srow = ((size_t)bt * V + v) * L + l;
  } else {
    const int v = r % V;
    r /= V;
    const int t2 = r % (2 * T);
    const int bb = r / (2 * T);
    e = t2 & 1;
    const int t = t2 >> 1;
    srow = (((size_t)bb * T + t) * V + v) * L + l;
  }
  const float* xp = src + srow * 512 + e * 256 + lane * 8;
  union { float4 f4[2]; float f[8]; } x, gg, bv;
  x.f4[0] = *(const float4*)(xp);
  x.f4[1] = *(const float4*)(xp + 4);
  float s = 0.f;
#pragma unroll
  for (int j = 0; j < 8; ++j) s += x.f[j];
#pragma unroll
  for (int os = 16; os > 0; os >>= 1) s += __shfl_xor(s, os, 32);
  const float mu = s * (1.f / 256.f);
  float vv = 0.f;
#pragma unroll
  for (int j = 0; j < 8; ++j) {
    const float d = x.f[j] - mu;
    vv += d * d;
  }
#pragma unroll
  for (int os = 16; os > 0; os >>= 1) vv += __shfl_xor(vv, os, 32);
  const float rstd = rsqrtf(vv * (1.f / 256.f) + 1e-5f);
  gg.f4[0] = *(const float4*)(g + lane * 8);
  gg.f4[1] = *(const float4*)(g + lane * 8 + 4);
  bv.f4[0] = *(const float4*)(b + lane * 8);
  bv.f4[1] = *(const float4*)(b + lane * 8 + 4);
  union { float4 f4[2]; float f[8]; } oo;
#pragma unroll
  for (int j = 0; j < 8; ++j) oo.f[j] = (x.f[j] - mu) * rstd * gg.f[j] + bv.f[j];
  float4* op = (float4*)(hout + (size_t)o * 256 + lane * 8);
  op[0] = oo.f4[0];
  op[1] = oo.f4[1];
}

__global__ void cvt_f32_bf16(const float* __restrict__ in, bf16t* __restrict__ out,
                             size_t n8) {  // n8 = n / 8
  size_t i = (size_t)blockIdx.x * blockDim.x + threadIdx.x;
  const size_t stride = (size_t)gridDim.x * blockDim.x;
  for (; i < n8; i += stride) {
    const float4 a = ((const float4*)in)[i * 2];
    const float4 b = ((const float4*)in)[i * 2 + 1];
    Pack8 p;
    p.h[0] = (bf16t)a.x; p.h[1] = (bf16t)a.y; p.h[2] = (bf16t)a.z; p.h[3] = (bf16t)a.w;
    p.h[4] = (bf16t)b.x; p.h[5] = (bf16t)b.y; p.h[6] = (bf16t)b.z; p.h[7] = (bf16t)b.w;
    ((uint4*)out)[i] = p.u;
  }
}

__global__ void add_inplace(float* __restrict__ h, const float* __restrict__ y,
                            size_t n4) {  // n4 = n / 4
  size_t i = (size_t)blockIdx.x * blockDim.x + threadIdx.x;
  const size_t stride = (size_t)gridDim.x * blockDim.x;
  for (; i < n4; i += stride) {
    float4 a = ((float4*)h)[i];
    const float4 b = ((const float4*)y)[i];
    a.x += b.x; a.y += b.y; a.z += b.z; a.w += b.w;
    ((float4*)h)[i] = a;
  }
}

// ---------------------------------------------------------------------------
extern "C" void kernel_launch(void* const* d_in, const int* in_sizes, int n_in,
                              void* d_out, int out_size, void* d_ws, size_t ws_size,
                              hipStream_t stream) {
  (void)in_sizes; (void)n_in; (void)out_size; (void)ws_size;
  const float* x      = (const float*)d_in[0];
  const float* n1w    = (const float*)d_in[1];
  const float* n1b    = (const float*)d_in[2];
  const float* qkv_w  = (const float*)d_in[3];
  const float* qkv_b  = (const float*)d_in[4];
  const float* rpb    = (const float*)d_in[5];
  const float* proj_w = (const float*)d_in[6];
  const float* proj_b = (const float*)d_in[7];
  const float* n2w    = (const float*)d_in[8];
  const float* n2b    = (const float*)d_in[9];
  const float* mlp1_w = (const float*)d_in[10];
  const float* mlp1_b = (const float*)d_in[11];
  const float* mlp2_w = (const float*)d_in[12];
  const float* mlp2_b = (const float*)d_in[13];
  const float* exp_v_w  = (const float*)d_in[14];
  const float* exp_v_nw = (const float*)d_in[15];
  const float* exp_v_nb = (const float*)d_in[16];
  const float* exp_t_w  = (const float*)d_in[17];
  const float* exp_t_nw = (const float*)d_in[18];
  const float* exp_t_nb = (const float*)d_in[19];

  // Workspace layout (MAXT = stage-3 tokens)
  const size_t MAXT = 131072;
  char* ws = (char*)d_ws;
  size_t off = 0;
  float* hbuf = (float*)(ws + off); off += MAXT * 256 * sizeof(float);   // stage 1/2 h
  bf16t* bufA = (bf16t*)(ws + off); off += MAXT * 256 * sizeof(bf16t);   // GEMM A ops
  bf16t* bufB = (bf16t*)(ws + off); off += MAXT * 768 * sizeof(bf16t);   // qkv
  bf16t* bufH = (bf16t*)(ws + off); off += MAXT * 1024 * sizeof(bf16t);  // mlp hidden
  float* bufC = (float*)(ws + off); off += MAXT * 512 * sizeof(float);   // f32 gemm out

  const dim3 blk(256);

  auto run_block = [&](float* h, int i, int D, int shifted) {
    const int tokens = 2 * D * 4096;  // B=2, L=4096
    const int nW = tokens / 128;
    const int sd = shifted ? 1 : 0, sh = shifted ? 4 : 0, sw = shifted ? 4 : 0;
    // LN1 + shift + window partition -> bf16
    ln_part<<<tokens / 8, blk, 0, stream>>>(h, n1w + i * 256, n1b + i * 256, bufA,
                                            1, D, sd, sh, sw);
    // QKV GEMM -> bf16
    gemm_bf16<<<dim3(768 / 64, tokens / 128), blk, 0, stream>>>(
        bufA, 256, qkv_w + (size_t)i * 256 * 768, 768, qkv_b + i * 768, nullptr,
        bufB, 768, 256, 1);
    // windowed attention -> bf16 (overwrites bufA)
    win_attn<<<nW * 4, blk, 0, stream>>>(bufB, rpb + (size_t)i * 675 * 4, bufA, D,
                                         shifted);
    // proj GEMM -> f32
    gemm_bf16<<<dim3(256 / 64, tokens / 128), blk, 0, stream>>>(
        bufA, 256, proj_w + (size_t)i * 256 * 256, 256, proj_b + i * 256, bufC,
        nullptr, 256, 256, 0);
    // window reverse + unshift + residual
    win_reverse_add<<<tokens / 8, blk, 0, stream>>>(bufC, h, D, sd, sh, sw);
    // LN2 -> bf16
    ln_part<<<tokens / 8, blk, 0, stream>>>(h, n2w + i * 256, n2b + i * 256, bufA,
                                            0, D, 0, 0, 0);
    // MLP1 + GELU -> bf16
    gemm_bf16<<<dim3(1024 / 64, tokens / 128), blk, 0, stream>>>(
        bufA, 256, mlp1_w + (size_t)i * 256 * 1024, 1024, mlp1_b + i * 1024,
        nullptr, bufH, 1024, 256, 2);
    // MLP2 -> f32
    gemm_bf16<<<dim3(256 / 64, tokens / 128), blk, 0, stream>>>(
        bufH, 1024, mlp2_w + (size_t)i * 1024 * 256, 256, mlp2_b + i * 256, bufC,
        nullptr, 256, 1024, 0);
    // residual
    add_inplace<<<2048, blk, 0, stream>>>(h, bufC, (size_t)tokens * 64);
  };

  // ---- stage 1: D = T*V = 4, tokens = 32768 ----
  hipMemcpyAsync(hbuf, x, (size_t)32768 * 256 * sizeof(float),
                 hipMemcpyDeviceToDevice, stream);
  run_block(hbuf, 0, 4, 0);
  run_block(hbuf, 1, 4, 1);

  // ---- view patch expand (V -> 2V) + LN ----
  cvt_f32_bf16<<<2048, blk, 0, stream>>>(hbuf, bufA, (size_t)32768 * 32);
  gemm_bf16<<<dim3(512 / 64, 32768 / 128), blk, 0, stream>>>(
      bufA, 256, exp_v_w, 512, nullptr, bufC, nullptr, 512, 256, 0);
  expand_ln<<<65536 / 8, blk, 0, stream>>>(bufC, exp_v_nw, exp_v_nb, hbuf,
                                           /*T=*/2, /*V=*/2, /*mode=*/0);

  // ---- stage 2: D = 8, tokens = 65536 ----
  run_block(hbuf, 2, 8, 0);
  run_block(hbuf, 3, 8, 1);

  // ---- temporal patch expand (T -> 2T) + LN, write into d_out ----
  cvt_f32_bf16<<<2048, blk, 0, stream>>>(hbuf, bufA, (size_t)65536 * 32);
  gemm_bf16<<<dim3(512 / 64, 65536 / 128), blk, 0, stream>>>(
      bufA, 256, exp_t_w, 512, nullptr, bufC, nullptr, 512, 256, 0);
  float* h3 = (float*)d_out;
  expand_ln<<<131072 / 8, blk, 0, stream>>>(bufC, exp_t_nw, exp_t_nb, h3,
                                            /*T=*/2, /*V=*/4, /*mode=*/1);

  // ---- stage 3: D = 16, tokens = 131072, operate in-place on d_out ----
  run_block(h3, 4, 16, 0);
  run_block(h3, 5, 16, 1);
}
